// SEGNN_47107201302530
// MI455X (gfx1250) — matrix operand order repaired and checked by reading, not compiled, not run on hardware
//
#include <hip/hip_runtime.h>
#include <math.h>

// ---------------------------------------------------------------------------
// SEGNN forward for MI455X (gfx1250, wave32).
//   * Edge TP->Linear stacks run on v_wmma_f32_16x16x32_f16 (16 edges / wave).
//   * A fragments staged in LDS ([16][96] f16 rows -> 2x ds_load_b128/lane).
//   * B fragments read from pre-transposed f16 weights in workspace
//     ([Npad][Kpad], K contiguous per lane -> 2x global 16B loads, L2-resident).
//   * segment_sum via global_atomic_add_f32.
// Input order: setup_inputs() insertion order for the 10 arrays, then `params`
// flattened in jax tree (sorted-key) order -- see indices in kernel_launch.
// ---------------------------------------------------------------------------

#define INV_SQRT3F 0.57735026918962576f

typedef __attribute__((ext_vector_type(16))) _Float16 v16h;
typedef __attribute__((ext_vector_type(8)))  _Float16 v8h;
typedef __attribute__((ext_vector_type(8)))  float    v8f;

constexpr int kNodes = 10000;
constexpr int kEdges = 160000;
constexpr int NW = 2;    // edge tiles (waves) per block
constexpr int FW = 96;   // f16 A-staging row width (halves)
constexpr int SW = 80;   // f32 scalar-feature row width
constexpr int VW = 24;   // f32 per-component vector row width

// ---------------- activations (match jax.nn.gelu approximate=True) ----------
__device__ __forceinline__ float gelu_f(float x) {
  float x3 = x * x * x;
  return 0.5f * x * (1.0f + tanhf(0.7978845608028654f * (x + 0.044715f * x3)));
}
__device__ __forceinline__ float sigmoid_f(float x) { return 1.0f / (1.0f + expf(-x)); }

// ---------------- WMMA fragment loaders -------------------------------------
// 16-bit A 16x32 layout: lane m (0-15) / m+16 hold row M=m; halves h<8 map to
// K = (lane>>4)*8 + h, halves h>=8 map to K = 16 + (lane>>4)*8 + (h-8).
__device__ __forceinline__ v16h ld_frag_a(const _Float16* fa, int kbase) {
  const int l = threadIdx.x & 31;
  const _Float16* p = fa + (l & 15) * FW + kbase + ((l >> 4) << 3);
  union { v16h v; v8h h[2]; } u;
  u.h[0] = *reinterpret_cast<const v8h*>(p);        // K = base+hi*8 + 0..7
  u.h[1] = *reinterpret_cast<const v8h*>(p + 16);   // K = base+hi*8 + 16..23
  return u.v;
}
// 16-bit B 32x16 layout: lane = column N (mod 16); halves h<8 map to
// K = (lane>>4)*8 + h, h>=8 to K = 16 + (lane>>4)*8 + (h-8) ([Npad][Kpad] storage).
__device__ __forceinline__ v16h ld_frag_b(const _Float16* wt, int Kp, int nt, int kt) {
  const int l = threadIdx.x & 31;
  const _Float16* p = wt + (nt * 16 + (l & 15)) * Kp + kt * 32 + ((l >> 4) << 3);
  union { v16h v; v8h h[2]; } u;
  u.h[0] = *reinterpret_cast<const v8h*>(p);
  u.h[1] = *reinterpret_cast<const v8h*>(p + 16);
  return u.v;
}

// D (f32 16x16): lane l holds column l&15, rows (l>>4)*8 + r.
__device__ __forceinline__ void mm_tile(const _Float16* fa, const _Float16* wt, int Kp,
                                        const float* bias, int Nact, int ntiles,
                                        float* out, int outw) {
  const int l = threadIdx.x & 31;
  const int n = l & 15, mb = (l >> 4) * 8;
  const int nkt = Kp >> 5;
  for (int nt = 0; nt < ntiles; ++nt) {
    v8f c = {};
    for (int kt = 0; kt < nkt; ++kt) {
      v16h a = ld_frag_a(fa, kt * 32);
      v16h b = ld_frag_b(wt, Kp, nt, kt);
      c = __builtin_amdgcn_wmma_f32_16x16x32_f16(false, a, false, b, (short)0, c,
                                                 false, false);
    }
    const int col = nt * 16 + n;
    if (col < Nact) {
      const float bv = bias ? bias[col] : 0.0f;
#pragma unroll
      for (int r = 0; r < 8; ++r) out[(mb + r) * outw + col] = c[r] + bv;
    }
  }
}

// ---------------- Clebsch-Gordan TP staging into f16 A rows ------------------
// tp_s = [ xs*eas (ns) , (xv . eav)/sqrt3 (nv) ], zero-padded to Kp.
__device__ __forceinline__ void build_tp_s(_Float16* fa, const float* sIn,
                                           const float* vIn, const float* eb,
                                           int ns, int nv, int Kp) {
  const int l = threadIdx.x & 31, m = l & 15, h = l >> 4;
  const float* em = eb + m * 4;
  const float e0 = em[0], e1 = em[1], e2 = em[2], es = em[3];
  const float* xs = sIn + m * SW;
  const float* xv = vIn + m * 3 * VW;
  for (int k = h; k < Kp; k += 2) {
    float val = 0.0f;
    if (k < ns) val = xs[k] * es;
    else if (k < ns + nv) {
      int c = k - ns;
      val = (xv[c] * e0 + xv[VW + c] * e1 + xv[2 * VW + c] * e2) * INV_SQRT3F;
    }
    fa[m * FW + k] = (_Float16)val;
  }
}
// tp_v (component d) = [ xs*eav_d (ns) , xv_d*eas (nv) ], zero-padded.
__device__ __forceinline__ void build_tp_v(_Float16* fa, const float* sIn,
                                           const float* vIn, const float* eb,
                                           int ns, int nv, int Kp, int d) {
  const int l = threadIdx.x & 31, m = l & 15, h = l >> 4;
  const float* em = eb + m * 4;
  const float ed = em[d], es = em[3];
  const float* xs = sIn + m * SW;
  const float* xv = vIn + m * 3 * VW + d * VW;
  for (int k = h; k < Kp; k += 2) {
    float val = 0.0f;
    if (k < ns) val = xs[k] * ed;
    else if (k < ns + nv) val = xv[k - ns] * es;
    fa[m * FW + k] = (_Float16)val;
  }
}

__device__ __forceinline__ void gate_act(float* s, float* v) {
  const int l = threadIdx.x & 31, m = l & 15, h = l >> 4;
  float* sm = s + m * SW;
  float* vm = v + m * 3 * VW;
  for (int j = h; j < 32; j += 2) sm[j] = gelu_f(sm[j]);
  for (int c = h; c < 10; c += 2) {
    const float g = sigmoid_f(sm[32 + c]);
    vm[c] *= g; vm[VW + c] *= g; vm[2 * VW + c] *= g;
  }
}

// One TP->Linear(->gate) layer. Inputs (sIn,vIn) preserved, outputs (sOut,vOut).
__device__ __forceinline__ void run_layer(_Float16* fa, const float* sIn,
                                          const float* vIn, float* sOut, float* vOut,
                                          const float* eb, int ns, int nv, int Kp,
                                          const _Float16* wS, const _Float16* wV,
                                          const float* bias, int Ns, int ntS,
                                          bool gated) {
  build_tp_s(fa, sIn, vIn, eb, ns, nv, Kp);
  __syncthreads();
  mm_tile(fa, wS, Kp, bias, Ns, ntS, sOut, SW);
  __syncthreads();
  for (int d = 0; d < 3; ++d) {
    build_tp_v(fa, sIn, vIn, eb, ns, nv, Kp, d);
    __syncthreads();
    mm_tile(fa, wV, Kp, nullptr, 10, 1, vOut + d * VW, 3 * VW);
    __syncthreads();
  }
  if (gated) { gate_act(sOut, vOut); __syncthreads(); }
}

// ---------------- edge message kernel (one 16-edge tile per wave) ------------
__global__ void __launch_bounds__(NW * 32)
k_edge(const int* __restrict__ senders, const int* __restrict__ receivers,
       const float* __restrict__ add_s, const float* __restrict__ add_v,
       const float* __restrict__ eas_g, const float* __restrict__ eav_g,
       const float* __restrict__ S, const float* __restrict__ V,
       float* __restrict__ aggS, float* __restrict__ aggV,
       const _Float16* __restrict__ w0s, const _Float16* __restrict__ w0v,
       const float* __restrict__ bs0,
       const _Float16* __restrict__ w1s, const _Float16* __restrict__ w1v,
       const float* __restrict__ bs1,
       const _Float16* __restrict__ w2s, const _Float16* __restrict__ w2v,
       const float* __restrict__ bs2) {
  __shared__ __align__(16) _Float16 sh_fa[NW * 16 * FW];
  __shared__ float sh_s[NW * 2 * 16 * SW];
  __shared__ float sh_v[NW * 2 * 16 * 3 * VW];
  __shared__ float sh_e[NW * 16 * 4];

  const int w = threadIdx.x >> 5;
  const int l = threadIdx.x & 31;
  const int m = l & 15, h = l >> 4;
  _Float16* fa = sh_fa + w * 16 * FW;
  float* s0 = sh_s + w * 2 * 16 * SW;      // ping
  float* s1 = s0 + 16 * SW;                // pong
  float* v0 = sh_v + w * 2 * 16 * 3 * VW;
  float* v1 = v0 + 16 * 3 * VW;
  float* eb = sh_e + w * 16 * 4;
  const int e = (blockIdx.x * NW + w) * 16 + m;

  // gather: lane-half 0 loads sender features, half 1 loads receiver features
  {
    const int node = (h == 0) ? senders[e] : receivers[e];
    float* xs = s0 + m * SW;
    float* xv = v0 + m * 3 * VW;
    for (int j = 0; j < 32; ++j) xs[1 + h * 32 + j] = S[node * 32 + j];
    for (int d = 0; d < 3; ++d) {
      const float* vp = V + node * 48 + d * 16;
      float* xd = xv + d * VW + 1 + h * 10;
      for (int c = 0; c < 10; ++c) xd[c] = vp[c];
    }
    if (h == 0) {
      xs[0] = add_s[e];
      for (int d = 0; d < 3; ++d) xv[d * VW] = add_v[e * 3 + d];
      float* em = eb + m * 4;
      em[0] = eav_g[e * 3 + 0]; em[1] = eav_g[e * 3 + 1];
      em[2] = eav_g[e * 3 + 2]; em[3] = eas_g[e];
    }
  }
  __syncthreads();

  // edge0: (65s,21v) K=86->96, out 42s+10v, gated
  run_layer(fa, s0, v0, s1, v1, eb, 65, 21, 96, w0s, w0v, bs0, 42, 3, true);
  // edge1: (32s,10v) K=42->64, out 42s+10v, gated
  run_layer(fa, s1, v1, s0, v0, eb, 32, 10, 64, w1s, w1v, bs1, 42, 3, true);
  // edge2: (32s,10v) K=42->64, out 32s+10v, ungated
  run_layer(fa, s0, v0, s1, v1, eb, 32, 10, 64, w2s, w2v, bs2, 32, 2, false);
  __syncthreads();

  // scatter (segment_sum over receivers)
  const int recv = receivers[e];
  const float* sm = s1 + m * SW;
  const float* vm = v1 + m * 3 * VW;
  for (int j = h * 16; j < h * 16 + 16; ++j) atomicAdd(&aggS[recv * 32 + j], sm[j]);
  for (int idx = h; idx < 30; idx += 2) {
    const int d = idx / 10, c = idx % 10;
    atomicAdd(&aggV[recv * 48 + d * 16 + c], vm[d * VW + c]);
  }
}

// ---------------- weight transpose/convert (f32 [K][N] -> f16 [Npad][Kpad]) --
__global__ void k_prep_w(const float* __restrict__ W, _Float16* __restrict__ wT,
                         int K, int N, int Kp, int Npad) {
  const int i = blockIdx.x * blockDim.x + threadIdx.x;
  if (i >= Npad * Kp) return;
  const int n = i / Kp, k = i % Kp;
  wT[i] = (_Float16)((n < N && k < K) ? W[k * N + n] : 0.0f);
}

__global__ void k_zero(float* __restrict__ p, int n) {
  const int i = blockIdx.x * blockDim.x + threadIdx.x;
  if (i < n) p[i] = 0.0f;
}

// ---------------- embed ------------------------------------------------------
__global__ void k_embed(const float* __restrict__ node_s, const float* __restrict__ node_v,
                        const float* __restrict__ nas_g, const float* __restrict__ nav_g,
                        const float* __restrict__ Ws, const float* __restrict__ Wv,
                        const float* __restrict__ bs,
                        float* __restrict__ S, float* __restrict__ V, int n) {
  const int i = blockIdx.x * blockDim.x + threadIdx.x;
  if (i >= n) return;
  const float xs = node_s[i], ys = nas_g[i];
  float xv[3], yv[3];
  for (int d = 0; d < 3; ++d) { xv[d] = node_v[i * 3 + d]; yv[d] = nav_g[i * 3 + d]; }
  const float t0 = xs * ys;
  const float t1 = (xv[0] * yv[0] + xv[1] * yv[1] + xv[2] * yv[2]) * INV_SQRT3F;
  for (int j = 0; j < 32; ++j) S[i * 32 + j] = bs[j] + t0 * Ws[j] + t1 * Ws[32 + j];
  for (int d = 0; d < 3; ++d) {
    const float a = xs * yv[d], b = xv[d] * ys;
    for (int c = 0; c < 10; ++c) V[i * 48 + d * 16 + c] = a * Wv[c] + b * Wv[10 + c];
  }
}

// ---------------- node update (TP->Linear, residual) -------------------------
__global__ void k_node(const float* __restrict__ nas_g, const float* __restrict__ nav_g,
                       const float* __restrict__ Ws, const float* __restrict__ Wv,
                       const float* __restrict__ bs,
                       const float* __restrict__ aggS, const float* __restrict__ aggV,
                       float* __restrict__ S, float* __restrict__ V, int n) {
  const int i = blockIdx.x * blockDim.x + threadIdx.x;
  if (i >= n) return;
  const float nas = nas_g[i];
  float nav[3];
  for (int d = 0; d < 3; ++d) nav[d] = nav_g[i * 3 + d];
  float cs[64], cv[3][20];
  for (int j = 0; j < 32; ++j) { cs[j] = S[i * 32 + j]; cs[32 + j] = aggS[i * 32 + j]; }
  for (int d = 0; d < 3; ++d)
    for (int c = 0; c < 10; ++c) {
      cv[d][c] = V[i * 48 + d * 16 + c];
      cv[d][10 + c] = aggV[i * 48 + d * 16 + c];
    }
  float os[32], ov[30];
  for (int j = 0; j < 32; ++j) os[j] = bs[j];
  for (int j = 0; j < 30; ++j) ov[j] = 0.0f;
  for (int k = 0; k < 84; ++k) {
    float tps, tpv[3];
    if (k < 64) {
      tps = cs[k] * nas;
      for (int d = 0; d < 3; ++d) tpv[d] = cs[k] * nav[d];
    } else {
      const int c = k - 64;
      tps = (cv[0][c] * nav[0] + cv[1][c] * nav[1] + cv[2][c] * nav[2]) * INV_SQRT3F;
      for (int d = 0; d < 3; ++d) tpv[d] = cv[d][c] * nas;
    }
    const float* wr = Ws + k * 32;
    for (int j = 0; j < 32; ++j) os[j] += tps * wr[j];
    const float* wv = Wv + k * 10;
    for (int c = 0; c < 10; ++c) {
      const float wk = wv[c];
      for (int d = 0; d < 3; ++d) ov[d * 10 + c] += tpv[d] * wk;
    }
  }
  for (int j = 0; j < 32; ++j) S[i * 32 + j] += os[j];
  for (int d = 0; d < 3; ++d)
    for (int c = 0; c < 10; ++c) V[i * 48 + d * 16 + c] += ov[d * 10 + c];
}

// ---------------- prepool + dense + mean-reduce ------------------------------
__global__ void k_prepool(const float* __restrict__ nas_g, const float* __restrict__ nav_g,
                          const float* __restrict__ Wps, const float* __restrict__ bps,
                          const float* __restrict__ Wd, const float* __restrict__ bd,
                          const float* __restrict__ S, const float* __restrict__ V,
                          float* __restrict__ hsum, int n) {
  __shared__ float hacc[64];
  const int t = threadIdx.x;
  if (t < 64) hacc[t] = 0.0f;
  __syncthreads();
  const int i = blockIdx.x * blockDim.x + t;
  if (i < n) {
    const float nas = nas_g[i];
    float nav[3];
    for (int d = 0; d < 3; ++d) nav[d] = nav_g[i * 3 + d];
    float tps[42];
    for (int j = 0; j < 32; ++j) tps[j] = S[i * 32 + j] * nas;
    for (int c = 0; c < 10; ++c)
      tps[32 + c] = (V[i * 48 + c] * nav[0] + V[i * 48 + 16 + c] * nav[1] +
                     V[i * 48 + 32 + c] * nav[2]) * INV_SQRT3F;
    float pre[64];
    for (int j = 0; j < 64; ++j) {
      float a = bps[j];
      for (int k = 0; k < 42; ++k) a += tps[k] * Wps[k * 64 + j];
      pre[j] = a;
    }
    for (int j = 0; j < 64; ++j) {
      float a = bd[j];
      for (int k = 0; k < 64; ++k) a += pre[k] * Wd[k * 64 + j];
      atomicAdd(&hacc[j], a);
    }
  }
  __syncthreads();
  if (t < 64) atomicAdd(&hsum[t], hacc[t]);
}

// ---------------- final MLP (single block) -----------------------------------
__global__ void k_mlp(const float* __restrict__ hsum,
                      const float* __restrict__ W0, const float* __restrict__ b0,
                      const float* __restrict__ W1, const float* __restrict__ b1,
                      const float* __restrict__ W2, const float* __restrict__ b2,
                      const float* __restrict__ W3, const float* __restrict__ b3,
                      float* __restrict__ out, int nNodes) {
  __shared__ float a0[64], a1[256], a2[128], a3[128];
  const int t = threadIdx.x;
  if (t < 64) a0[t] = hsum[t] / (float)nNodes;
  __syncthreads();
  { float a = b0[t]; for (int k = 0; k < 64; ++k) a += a0[k] * W0[k * 256 + t]; a1[t] = gelu_f(a); }
  __syncthreads();
  if (t < 128) { float a = b1[t]; for (int k = 0; k < 256; ++k) a += a1[k] * W1[k * 128 + t]; a2[t] = gelu_f(a); }
  __syncthreads();
  if (t < 128) { float a = b2[t]; for (int k = 0; k < 128; ++k) a += a2[k] * W2[k * 128 + t]; a3[t] = gelu_f(a); }
  __syncthreads();
  if (t == 0) { float a = b3[0]; for (int k = 0; k < 128; ++k) a += a3[k] * W3[k]; out[0] = a; }
}

// ---------------------------------------------------------------------------
extern "C" void kernel_launch(void* const* d_in, const int* in_sizes, int n_in,
                              void* d_out, int out_size, void* d_ws, size_t ws_size,
                              hipStream_t stream) {
  (void)in_sizes; (void)n_in; (void)out_size; (void)ws_size;
  const float* node_s = (const float*)d_in[0];
  const float* node_v = (const float*)d_in[1];
  const float* nas    = (const float*)d_in[2];
  const float* nav    = (const float*)d_in[3];
  const float* eas    = (const float*)d_in[4];
  const float* eav    = (const float*)d_in[5];
  const float* add_s  = (const float*)d_in[6];
  const float* add_v  = (const float*)d_in[7];
  const int*   snd    = (const int*)d_in[8];
  const int*   rcv    = (const int*)d_in[9];
  // params, jax sorted-key tree order:
  const float* denseW = (const float*)d_in[10];  // (64,64)
  const float* denseB = (const float*)d_in[11];
  const float* embWs  = (const float*)d_in[12];  // (2,32)
  const float* embWv  = (const float*)d_in[13];  // (2,10)
  const float* embBs  = (const float*)d_in[14];
  const float* mlpW0  = (const float*)d_in[15];  // (64,256)
  const float* mlpB0  = (const float*)d_in[16];
  const float* mlpW1  = (const float*)d_in[17];  // (256,128)
  const float* mlpB1  = (const float*)d_in[18];
  const float* mlpW2  = (const float*)d_in[19];  // (128,128)
  const float* mlpB2  = (const float*)d_in[20];
  const float* mlpW3  = (const float*)d_in[21];  // (128,1)
  const float* mlpB3  = (const float*)d_in[22];
  const float* ppWs   = (const float*)d_in[23];  // (42,64)
  const float* ppBs   = (const float*)d_in[24];
  // steps: base 25 + 12*s : e0.Ws e0.Wv e0.bs e1.Ws e1.Wv e1.bs e2.Ws e2.Wv e2.bs nd.Ws nd.Wv nd.bs

  char* ws = (char*)d_ws;
  float* S    = (float*)(ws);                     // [10000][32]
  float* V    = (float*)(ws + 1280000);           // [10000][48] (10 used + pad)
  float* aggS = (float*)(ws + 3200000);           // [10000][32]
  float* aggV = (float*)(ws + 4480000);           // [10000][48]
  float* hsum = (float*)(ws + 6400000);           // [64]
  _Float16* wtb = (_Float16*)(ws + 6400256);      // f16 transposed weights, 3 steps

  // f16 weight prep (L2-resident thereafter)
  for (int s = 0; s < 3; ++s) {
    const int base = 25 + 12 * s;
    _Float16* wb = wtb + s * 13312;
    struct { int idx; int off; int K, N, Kp, Np; } mats[6] = {
      { base + 0, 0,     86, 42, 96, 48 },  // edge0.Ws
      { base + 1, 4608,  86, 10, 96, 16 },  // edge0.Wv
      { base + 3, 6144,  42, 42, 64, 48 },  // edge1.Ws
      { base + 4, 9216,  42, 10, 64, 16 },  // edge1.Wv
      { base + 6, 10240, 42, 32, 64, 32 },  // edge2.Ws
      { base + 7, 12288, 42, 10, 64, 16 },  // edge2.Wv
    };
    for (int j = 0; j < 6; ++j) {
      const int total = mats[j].Np * mats[j].Kp;
      k_prep_w<<<(total + 255) / 256, 256, 0, stream>>>(
          (const float*)d_in[mats[j].idx], wb + mats[j].off,
          mats[j].K, mats[j].N, mats[j].Kp, mats[j].Np);
    }
  }

  k_embed<<<(kNodes + 255) / 256, 256, 0, stream>>>(node_s, node_v, nas, nav,
                                                    embWs, embWv, embBs, S, V, kNodes);

  for (int s = 0; s < 3; ++s) {
    const int base = 25 + 12 * s;
    _Float16* wb = wtb + s * 13312;
    // zero aggS+aggV (contiguous, 800000 floats)
    k_zero<<<(800000 + 255) / 256, 256, 0, stream>>>(aggS, 800000);
    k_edge<<<kEdges / (NW * 16), NW * 32, 0, stream>>>(
        snd, rcv, add_s, add_v, eas, eav, S, V, aggS, aggV,
        wb + 0,     wb + 4608,  (const float*)d_in[base + 2],
        wb + 6144,  wb + 9216,  (const float*)d_in[base + 5],
        wb + 10240, wb + 12288, (const float*)d_in[base + 8]);
    k_node<<<(kNodes + 255) / 256, 256, 0, stream>>>(
        nas, nav, (const float*)d_in[base + 9], (const float*)d_in[base + 10],
        (const float*)d_in[base + 11], aggS, aggV, S, V, kNodes);
  }

  k_zero<<<1, 64, 0, stream>>>(hsum, 64);
  k_prepool<<<(kNodes + 255) / 256, 256, 0, stream>>>(nas, nav, ppWs, ppBs,
                                                      denseW, denseB, S, V, hsum, kNodes);
  k_mlp<<<1, 256, 0, stream>>>(hsum, mlpW0, mlpB0, mlpW1, mlpB1, mlpW2, mlpB2,
                               mlpW3, mlpB3, (float*)d_out, kNodes);
}
